// CURAttention_90357521973770
// MI455X (gfx1250) — matrix operand order repaired and checked by reading, not compiled, not run on hardware
//
#include <hip/hip_runtime.h>
#include <hip/hip_bf16.h>
#include <math.h>

// ---------------------------------------------------------------------------
// Problem constants (from reference): B=4 H=12 N=8192 D=64, M=128 landmarks
// ---------------------------------------------------------------------------
constexpr int B_  = 4;
constexpr int H_  = 12;
constexpr int N_  = 8192;
constexpr int D_  = 64;
constexpr int M_  = 128;
constexpr int BHc = B_ * H_;   // 48
constexpr int S_  = 8;         // N-slices for the flash-streaming kernel

typedef __attribute__((ext_vector_type(16))) _Float16 v16h;
typedef __attribute__((ext_vector_type(8)))  float    v8f;
typedef __attribute__((ext_vector_type(4)))  unsigned int v4u;
typedef __attribute__((ext_vector_type(8)))  int    v8i_;
typedef __attribute__((ext_vector_type(4)))  int    v4i_;

// ---------------------------------------------------------------------------
// WMMA helpers (CDNA5 16x16x32 f16 -> f32, wave32)
// ---------------------------------------------------------------------------
__device__ inline v8f v8f_zero() {
  v8f z;
#pragma unroll
  for (int i = 0; i < 8; ++i) z[i] = 0.0f;
  return z;
}

__device__ inline v8f wmma_f16(v16h a, v16h b, v8f c) {
  return __builtin_amdgcn_wmma_f32_16x16x32_f16(false, a, false, b, (short)0, c,
                                                false, false);
}

// A fragment: 16x32 f16 tile from row-major f16 [rows][lda].
__device__ inline v16h load_a16(const _Float16* __restrict__ A, int lda,
                                int row0, int k0) {
  const int lane = threadIdx.x & 31;
  const int grp  = lane >> 4;
  const int r    = row0 + (lane & 15);
  v16h f;
#pragma unroll
  for (int e = 0; e < 16; ++e) {
    const int k = e + grp * 16;
    f[e] = A[r * lda + k0 + k];
  }
  return f;
}

// A fragment from row-major f32 (convert at fragment load).
__device__ inline v16h load_a16_f32(const float* __restrict__ A, int lda,
                                    int row0, int k0) {
  const int lane = threadIdx.x & 31;
  const int grp  = lane >> 4;
  const int r    = row0 + (lane & 15);
  v16h f;
#pragma unroll
  for (int e = 0; e < 16; ++e) {
    const int k = e + grp * 16;
    f[e] = (_Float16)A[r * lda + k0 + k];
  }
  return f;
}

// B fragment: 32x16, B row-major f16 [K][ldb].
__device__ inline v16h load_b_row(const _Float16* __restrict__ Bm, int ldb,
                                  int k0, int n0) {
  const int lane = threadIdx.x & 31;
  const int grp  = lane >> 4;
  const int c    = n0 + (lane & 15);
  v16h f;
#pragma unroll
  for (int e = 0; e < 16; ++e) {
    const int k = k0 + e + grp * 16;
    f[e] = Bm[k * ldb + c];
  }
  return f;
}

// B fragment from row-major f32 [K][ldb] with convert.
__device__ inline v16h load_b_row_f32(const float* __restrict__ Bm, int ldb,
                                      int k0, int n0) {
  const int lane = threadIdx.x & 31;
  const int grp  = lane >> 4;
  const int c    = n0 + (lane & 15);
  v16h f;
#pragma unroll
  for (int e = 0; e < 16; ++e) {
    const int k = k0 + e + grp * 16;
    f[e] = (_Float16)Bm[k * ldb + c];
  }
  return f;
}

// B fragment for X @ Y^T: logical B[k][n] = Yt[n][ldt]+k, f16.
__device__ inline v16h load_b_col(const _Float16* __restrict__ Bt, int ldt,
                                  int k0, int n0) {
  const int lane = threadIdx.x & 31;
  const int grp  = lane >> 4;
  const int c    = n0 + (lane & 15);
  v16h f;
#pragma unroll
  for (int e = 0; e < 16; ++e) {
    const int k = k0 + e + grp * 16;
    f[e] = Bt[c * ldt + k];
  }
  return f;
}

// Same from f32 with convert.
__device__ inline v16h load_b_col_f32(const float* __restrict__ Bt, int ldt,
                                      int k0, int n0) {
  const int lane = threadIdx.x & 31;
  const int grp  = lane >> 4;
  const int c    = n0 + (lane & 15);
  v16h f;
#pragma unroll
  for (int e = 0; e < 16; ++e) {
    const int k = k0 + e + grp * 16;
    f[e] = (_Float16)Bt[c * ldt + k];
  }
  return f;
}

// B fragment of (alpha*I - Bm), f16 row-major 128x128 (Newton-Schulz step).
__device__ inline v16h load_b_diag(const _Float16* __restrict__ Bm, int ldb,
                                   int k0, int n0, float alpha) {
  const int lane = threadIdx.x & 31;
  const int grp  = lane >> 4;
  const int c    = n0 + (lane & 15);
  v16h f;
#pragma unroll
  for (int e = 0; e < 16; ++e) {
    const int k = k0 + e + grp * 16;
    float v = -(float)Bm[k * ldb + c];
    if (k == c) v += alpha;
    f[e] = (_Float16)v;
  }
  return f;
}

// 128x128 @ 128x128 matmul entirely in LDS; buffers selected by integer
// element offsets off a single LDS base so address-space inference stays
// intact (ds_ ops, no flat).  Optional B := (alpha*I - B) and output scale.
__device__ inline void mm128(_Float16* __restrict__ lds, int dOff, int aOff,
                             int bOff, int diag, float alpha, float scale) {
  const int wave = threadIdx.x >> 5;
  const int lane = threadIdx.x & 31;
  const int grp  = lane >> 4;
  const int r0   = wave * 16;
  __syncthreads();
  for (int ct = 0; ct < 8; ++ct) {
    v8f acc = v8f_zero();
#pragma unroll
    for (int kk = 0; kk < 4; ++kk) {
      const int k0 = kk * 32;
      v16h a = load_a16(lds + aOff, 128, r0, k0);
      v16h b = diag ? load_b_diag(lds + bOff, 128, k0, ct * 16, alpha)
                    : load_b_row(lds + bOff, 128, k0, ct * 16);
      acc = wmma_f16(a, b, acc);
    }
#pragma unroll
    for (int v = 0; v < 8; ++v) {
      const int row = r0 + v + grp * 8;
      const int col = ct * 16 + (lane & 15);
      lds[dOff + row * 128 + col] = (_Float16)(acc[v] * scale);
    }
  }
  __syncthreads();
}

// ---------------------------------------------------------------------------
// Tensor Data Mover: 2-D tile (f32, 4B elements) global -> LDS.
// D# per CDNA5 ISA ch.8: group0 = {count, lds_addr, global_addr, type=2},
// group1 = {data_size, tensor_dim0/1, tile_dim0/1, tensor_dim0_stride}.
// This toolchain exposes the 6-arg builtin:
//   (uint32x4 g0, int32x8 g1, int32x4 g2, int32x4 g3, int32x8 g4, i32 cpol)
// ---------------------------------------------------------------------------
__device__ inline unsigned lds_off_bytes(const void* p) {
  // LDS aperture: low 32 bits of the generic address are the LDS byte offset.
  return (unsigned)(unsigned long long)(uintptr_t)p;
}

__device__ inline void tdm_load_2d_f32(unsigned lds_byte, const void* gptr,
                                       unsigned tile_w, unsigned tile_h,
                                       unsigned stride_w, unsigned tens_w,
                                       unsigned tens_h) {
#if __has_builtin(__builtin_amdgcn_tensor_load_to_lds)
  const unsigned long long ga = (unsigned long long)(uintptr_t)gptr;
  v4u g0;
  g0[0] = 1u;                                   // count=1, user descriptor
  g0[1] = lds_byte;                             // lds_addr
  g0[2] = (unsigned)ga;                         // global_addr[31:0]
  g0[3] = (unsigned)((ga >> 32) & 0x01ffffffu)  // global_addr[56:32]
          | (2u << 30);                         // type = 2 ("image")
  // group1 (256 bits) assembled as 4x u64
  unsigned long long w0 = ((unsigned long long)2u << 16)                 // 4B
                        | ((unsigned long long)(tens_w & 0xffffu) << 48);
  unsigned long long w1 = ((unsigned long long)(tens_w >> 16) & 0xffffu)
                        | ((unsigned long long)tens_h << 16)
                        | ((unsigned long long)(tile_w & 0xffffu) << 48);
  unsigned long long w2 = (unsigned long long)(tile_h & 0xffffu)
                        | ((unsigned long long)stride_w << 32);
  unsigned long long w3 = 0ull;                 // dim1 stride unused (2D tile)
  v8i_ g1;
  g1[0] = (int)(unsigned)w0; g1[1] = (int)(unsigned)(w0 >> 32);
  g1[2] = (int)(unsigned)w1; g1[3] = (int)(unsigned)(w1 >> 32);
  g1[4] = (int)(unsigned)w2; g1[5] = (int)(unsigned)(w2 >> 32);
  g1[6] = (int)(unsigned)w3; g1[7] = (int)(unsigned)(w3 >> 32);
  v4i_ gz4 = {0, 0, 0, 0};
  v8i_ gz8 = {0, 0, 0, 0, 0, 0, 0, 0};
  __builtin_amdgcn_tensor_load_to_lds(g0, g1, gz4, gz4, gz8, 0);
#else
  (void)lds_byte; (void)gptr; (void)tile_w; (void)tile_h;
  (void)stride_w; (void)tens_w; (void)tens_h;
#endif
}

__device__ inline void tdm_wait() {
#if __has_builtin(__builtin_amdgcn_tensor_load_to_lds)
  __builtin_amdgcn_s_wait_tensorcnt(0);
#endif
}

// ---------------------------------------------------------------------------
// Kernel 1: landmark selection.  grid=(BH,2): y==0 -> K, y==1 -> Q.
// ---------------------------------------------------------------------------
__global__ void topk_kernel(const float* __restrict__ Kg,
                            const float* __restrict__ Qg,
                            int* __restrict__ idxK, int* __restrict__ idxQ) {
  __shared__ float vals[N_];
  __shared__ float rv[256];
  __shared__ int   ri[256];
  __shared__ int   sel[M_];
  const int bh  = blockIdx.x;
  const int tid = threadIdx.x;
  const float* src = (blockIdx.y == 0) ? Kg : Qg;
  int*         dst = (blockIdx.y == 0) ? idxK : idxQ;

  for (int i = tid; i < N_; i += 256)
    vals[i] = src[(size_t)bh * N_ * D_ + (size_t)i * D_];
  __syncthreads();
  if (tid == 0) { vals[0] = -1e30f; sel[0] = 0; }
  __syncthreads();

  for (int it = 1; it < M_; ++it) {
    float bv = -1e30f; int bi = 0;
    for (int i = tid; i < N_; i += 256) {
      const float v = vals[i];
      if (v > bv) { bv = v; bi = i; }
    }
    rv[tid] = bv; ri[tid] = bi;
    __syncthreads();
    for (int sd = 128; sd > 0; sd >>= 1) {
      if (tid < sd) {
        if (rv[tid + sd] > rv[tid] ||
            (rv[tid + sd] == rv[tid] && ri[tid + sd] < ri[tid])) {
          rv[tid] = rv[tid + sd]; ri[tid] = ri[tid + sd];
        }
      }
      __syncthreads();
    }
    if (tid == 0) { sel[it] = ri[0]; vals[ri[0]] = -1e30f; }
    __syncthreads();
  }
  for (int p = 0; p < M_; ++p) {
    const int i = 2 * tid + (p & 1);
    if (tid < 64 && i + 1 < M_) {
      const int a = sel[i], b = sel[i + 1];
      if (a > b) { sel[i] = b; sel[i + 1] = a; }
    }
    __syncthreads();
  }
  if (tid < M_) dst[(size_t)bh * M_ + tid] = sel[tid];
}

// ---------------------------------------------------------------------------
// Kernel 2: gather landmark rows to f16.  grid=BH, block=256.
// ---------------------------------------------------------------------------
__global__ void gather_kernel(const float* __restrict__ Kg,
                              const float* __restrict__ Qg,
                              const int* __restrict__ idxK,
                              const int* __restrict__ idxQ,
                              _Float16* __restrict__ nc16,
                              _Float16* __restrict__ nr16) {
  const int bh = blockIdx.x, tid = threadIdx.x;
  const int row = tid >> 1, half = tid & 1;
  const int ik = idxK[(size_t)bh * M_ + row];
  const int iq = idxQ[(size_t)bh * M_ + row];
  const size_t gb = (size_t)bh * N_ * D_;
  const size_t lb = (size_t)bh * M_ * D_;
  for (int c = half * 32; c < half * 32 + 32; ++c) {
    nc16[lb + row * D_ + c] = (_Float16)Kg[gb + (size_t)ik * D_ + c];
    nr16[lb + row * D_ + c] = (_Float16)Qg[gb + (size_t)iq * D_ + c];
  }
}

// ---------------------------------------------------------------------------
// Kernel 3: m = nr@nc^T -> softmax -> 6x Newton-Schulz -> kernel_2^{-1}.
// grid=BH, block=256.  All LDS buffers addressed by integer offsets.
// ---------------------------------------------------------------------------
__global__ void inv_kernel(const _Float16* __restrict__ nr16,
                           const _Float16* __restrict__ nc16,
                           _Float16* __restrict__ k2i16) {
  extern __shared__ _Float16 lds16[];
  __shared__ float scs[M_];
  __shared__ float denom;
  // element offsets of the five 128x128 f16 buffers
  const int oK = 0, oP0 = 16384, oP1 = 32768, oP2 = 49152, oP3 = 65536;

  const int bh = blockIdx.x, tid = threadIdx.x;
  const int wave = tid >> 5, lane = tid & 31, grp = lane >> 4;
  const size_t lb = (size_t)bh * M_ * D_;

  for (int i = tid; i < M_ * D_; i += 256) {
    lds16[oP0 + i] = nr16[lb + i];
    lds16[oP1 + i] = nc16[lb + i];
  }
  __syncthreads();

  // m = nr @ nc^T with row softmax in registers -> kernel_2 at oK
  {
    const int r0 = wave * 16;
    v8f acc[8];
    for (int ct = 0; ct < 8; ++ct) {
      v8f a_ = v8f_zero();
#pragma unroll
      for (int kk = 0; kk < 2; ++kk) {
        v16h a = load_a16(lds16 + oP0, D_, r0, kk * 32);
        v16h b = load_b_col(lds16 + oP1, D_, kk * 32, ct * 16);
        a_ = wmma_f16(a, b, a_);
      }
      acc[ct] = a_;
    }
#pragma unroll
    for (int v = 0; v < 8; ++v) {
      float mx = -1e30f;
#pragma unroll
      for (int ct = 0; ct < 8; ++ct) mx = fmaxf(mx, acc[ct][v]);
      mx = fmaxf(mx, __shfl_xor(mx, 1));
      mx = fmaxf(mx, __shfl_xor(mx, 2));
      mx = fmaxf(mx, __shfl_xor(mx, 4));
      mx = fmaxf(mx, __shfl_xor(mx, 8));
      float s = 0.0f;
#pragma unroll
      for (int ct = 0; ct < 8; ++ct) {
        const float e = __expf(acc[ct][v] - mx);
        acc[ct][v] = e; s += e;
      }
      s += __shfl_xor(s, 1); s += __shfl_xor(s, 2);
      s += __shfl_xor(s, 4); s += __shfl_xor(s, 8);
      const float inv = 1.0f / s;
#pragma unroll
      for (int ct = 0; ct < 8; ++ct) acc[ct][v] *= inv;
    }
    for (int ct = 0; ct < 8; ++ct)
#pragma unroll
      for (int v = 0; v < 8; ++v)
        lds16[oK + (r0 + v + grp * 8) * 128 + ct * 16 + (lane & 15)] =
            (_Float16)acc[ct][v];
  }
  __syncthreads();

  // Vm0 = K2^T / max(colsum(K2))
  if (tid < 128) {
    float s = 0.0f;
    for (int r = 0; r < 128; ++r) s += (float)lds16[oK + r * 128 + tid];
    scs[tid] = s;
  }
  __syncthreads();
  if (tid == 0) {
    float mx = scs[0];
    for (int i = 1; i < 128; ++i) mx = fmaxf(mx, scs[i]);
    denom = mx;
  }
  __syncthreads();
  for (int i = tid; i < 128 * 128; i += 256) {
    const int r = i >> 7, c = i & 127;
    lds16[oP0 + i] = (_Float16)((float)lds16[oK + c * 128 + r] / denom);
  }

  // 6 Newton-Schulz iterations; buffer roles tracked as integer offsets.
  int vOff = oP0, qOff = oP1;
  for (int it = 0; it < 6; ++it) {
    mm128(lds16, qOff, oK,  vOff, 0, 0.0f,  1.0f);   // KV
    mm128(lds16, oP2,  qOff, qOff, 1, 7.0f,  1.0f);  // T2 = KV (7I - KV)
    mm128(lds16, oP3,  qOff, oP2,  1, 15.0f, 1.0f);  // T3 = KV (15I - T2)
    mm128(lds16, qOff, vOff, oP3,  1, 13.0f, 0.25f); // Vnew (over dead KV)
    const int t = vOff; vOff = qOff; qOff = t;
  }
  for (int i = tid; i < M_ * M_; i += 256)
    k2i16[(size_t)bh * M_ * M_ + i] = lds16[vOff + i];
}

// ---------------------------------------------------------------------------
// Kernel 4: flash streaming r = nr@K^T, online softmax, A3V += P@V.
// K/V tiles come in through the Tensor Data Mover (f32 in LDS, converted at
// fragment-load time).  grid=(S_, BH), block=256.
// ---------------------------------------------------------------------------
__global__ void rkv_kernel(const float* __restrict__ Kg,
                           const float* __restrict__ Vg,
                           const _Float16* __restrict__ nr16,
                           float* __restrict__ a3v_part,
                           float* __restrict__ m_part,
                           float* __restrict__ l_part) {
  extern __shared__ _Float16 lds16[];
  float*     sKf = (float*)lds16;            // 64x64 f32  (16 KB)
  float*     sVf = (float*)(lds16 + 8192);   // 64x64 f32  (16 KB)
  _Float16*  sNR = lds16 + 16384;            // 128x64 f16 (16 KB)
  _Float16*  sP  = lds16 + 24576;            // 8 waves * 16x64 f16 (16 KB)

  const int s = blockIdx.x, bh = blockIdx.y;
  const int tid = threadIdx.x, wave = tid >> 5, lane = tid & 31, grp = lane >> 4;
  const size_t base = (size_t)bh * N_ * D_;
  const unsigned kOff = lds_off_bytes(sKf);
  const unsigned vOff = lds_off_bytes(sVf);

  for (int i = tid; i < M_ * D_; i += 256) sNR[i] = nr16[(size_t)bh * M_ * D_ + i];

  float run_m[8], run_l[8];
  v8f oacc[4];
#pragma unroll
  for (int v = 0; v < 8; ++v) { run_m[v] = -1e30f; run_l[v] = 0.0f; }
  for (int dt = 0; dt < 4; ++dt) oacc[dt] = v8f_zero();

  const int n0s = s * (N_ / S_);
  const int nchunks = (N_ / S_) / 64;
  for (int ch = 0; ch < nchunks; ++ch) {
    const int n0 = n0s + ch * 64;
    __syncthreads();   // previous chunk compute done before DMA overwrites
    if (wave == 0) {   // one TDM DMA per tile (TDM ignores EXEC; per-wave op)
      tdm_load_2d_f32(kOff, Kg + base + (size_t)n0 * D_, D_, 64, D_, D_, N_);
      tdm_load_2d_f32(vOff, Vg + base + (size_t)n0 * D_, D_, 64, D_, D_, N_);
      tdm_wait();
    }
    __syncthreads();

    const int r0 = wave * 16;
    v8f racc[4];
    for (int ct = 0; ct < 4; ++ct) {
      v8f a_ = v8f_zero();
#pragma unroll
      for (int kk = 0; kk < 2; ++kk) {
        v16h a = load_a16(sNR, D_, r0, kk * 32);
        v16h b = load_b_col_f32(sKf, D_, kk * 32, ct * 16);
        a_ = wmma_f16(a, b, a_);
      }
      racc[ct] = a_;
    }
#pragma unroll
    for (int v = 0; v < 8; ++v) {
      float mx = -1e30f;
#pragma unroll
      for (int ct = 0; ct < 4; ++ct) mx = fmaxf(mx, racc[ct][v]);
      mx = fmaxf(mx, __shfl_xor(mx, 1));
      mx = fmaxf(mx, __shfl_xor(mx, 2));
      mx = fmaxf(mx, __shfl_xor(mx, 4));
      mx = fmaxf(mx, __shfl_xor(mx, 8));
      const float nm = fmaxf(run_m[v], mx);
      const float sc = __expf(run_m[v] - nm);
      float sum = 0.0f;
#pragma unroll
      for (int ct = 0; ct < 4; ++ct) {
        const float e = __expf(racc[ct][v] - nm);
        racc[ct][v] = e; sum += e;
      }
      sum += __shfl_xor(sum, 1); sum += __shfl_xor(sum, 2);
      sum += __shfl_xor(sum, 4); sum += __shfl_xor(sum, 8);
      run_l[v] = run_l[v] * sc + sum;
      run_m[v] = nm;
#pragma unroll
      for (int dt = 0; dt < 4; ++dt) oacc[dt][v] *= sc;
    }
    _Float16* sPw = sP + wave * 16 * 64;
    for (int ct = 0; ct < 4; ++ct)
#pragma unroll
      for (int v = 0; v < 8; ++v)
        sPw[(v + grp * 8) * 64 + ct * 16 + (lane & 15)] = (_Float16)racc[ct][v];
    __syncthreads();
    for (int dt = 0; dt < 4; ++dt) {
      v8f a_ = oacc[dt];
#pragma unroll
      for (int kk = 0; kk < 2; ++kk) {
        v16h a = load_a16(sPw, D_, 0, kk * 32);
        v16h b = load_b_row_f32(sVf, D_, kk * 32, dt * 16);
        a_ = wmma_f16(a, b, a_);
      }
      oacc[dt] = a_;
    }
  }

  const size_t pbase = (size_t)bh * S_ + s;
  float* A3 = a3v_part + pbase * M_ * D_;
  for (int dt = 0; dt < 4; ++dt)
#pragma unroll
    for (int v = 0; v < 8; ++v)
      A3[(wave * 16 + v + grp * 8) * D_ + dt * 16 + (lane & 15)] = oacc[dt][v];
  if ((lane & 15) == 0) {
#pragma unroll
    for (int v = 0; v < 8; ++v) {
      const int row = wave * 16 + v + grp * 8;
      m_part[pbase * M_ + row] = run_m[v];
      l_part[pbase * M_ + row] = run_l[v];
    }
  }
}

// ---------------------------------------------------------------------------
// Kernel 4b: merge the S_ flash partials -> a3v16 (f16). grid=BH, block=256.
// ---------------------------------------------------------------------------
__global__ void merge_kernel(const float* __restrict__ a3v_part,
                             const float* __restrict__ m_part,
                             const float* __restrict__ l_part,
                             _Float16* __restrict__ a3v16) {
  __shared__ float wM[M_], wL[M_];
  const int bh = blockIdx.x, tid = threadIdx.x;
  if (tid < M_) {
    float mx = -1e30f;
    for (int s = 0; s < S_; ++s)
      mx = fmaxf(mx, m_part[((size_t)bh * S_ + s) * M_ + tid]);
    float L = 0.0f;
    for (int s = 0; s < S_; ++s)
      L += l_part[((size_t)bh * S_ + s) * M_ + tid] *
           __expf(m_part[((size_t)bh * S_ + s) * M_ + tid] - mx);
    wM[tid] = mx; wL[tid] = L;
  }
  __syncthreads();
  for (int i = tid; i < M_ * D_; i += 256) {
    const int row = i >> 6;
    float acc = 0.0f;
    for (int s = 0; s < S_; ++s)
      acc += a3v_part[((size_t)bh * S_ + s) * M_ * D_ + i] *
             __expf(m_part[((size_t)bh * S_ + s) * M_ + row] - wM[row]);
    a3v16[(size_t)bh * M_ * D_ + i] = (_Float16)(acc / wL[row]);
  }
}

// ---------------------------------------------------------------------------
// Kernel 5: W = kernel_2^{-1} @ (kernel_3 @ V).  grid=BH, block=256.
// ---------------------------------------------------------------------------
__global__ void wmat_kernel(const _Float16* __restrict__ k2i16,
                            const _Float16* __restrict__ a3v16,
                            _Float16* __restrict__ w16) {
  extern __shared__ _Float16 lds16[];
  const int oA = 0, oB = 16384;  // 128x128 f16, 128x64 f16
  const int bh = blockIdx.x, tid = threadIdx.x;
  const int wave = tid >> 5, lane = tid & 31, grp = lane >> 4;
  for (int i = tid; i < M_ * M_; i += 256)
    lds16[oA + i] = k2i16[(size_t)bh * M_ * M_ + i];
  for (int i = tid; i < M_ * D_; i += 256)
    lds16[oB + i] = a3v16[(size_t)bh * M_ * D_ + i];
  __syncthreads();
  const int r0 = wave * 16;
  for (int dt = 0; dt < 4; ++dt) {
    v8f acc = v8f_zero();
#pragma unroll
    for (int kk = 0; kk < 4; ++kk) {
      v16h a = load_a16(lds16 + oA, M_, r0, kk * 32);
      v16h b = load_b_row(lds16 + oB, D_, kk * 32, dt * 16);
      acc = wmma_f16(a, b, acc);
    }
#pragma unroll
    for (int v = 0; v < 8; ++v)
      w16[(size_t)bh * M_ * D_ + (r0 + v + grp * 8) * D_ + dt * 16 + (lane & 15)] =
          (_Float16)acc[v];
  }
}

// ---------------------------------------------------------------------------
// Kernel 6: stream Q (via TDM): c = Qtile@nc^T, softmax, out = P @ W.
// grid=(N/128, BH), block=256.
// ---------------------------------------------------------------------------
__global__ void out_kernel(const float* __restrict__ Qg,
                           const _Float16* __restrict__ nc16,
                           const _Float16* __restrict__ w16,
                           float* __restrict__ Out) {
  extern __shared__ _Float16 lds16[];
  float*    sQf = (float*)lds16;            // 128x64 f32 (32 KB)
  _Float16* sNC = lds16 + 16384;            // 128x64 f16 (16 KB)
  _Float16* sW  = lds16 + 24576;            // 128x64 f16 (16 KB)
  _Float16* sP  = lds16 + 32768;            // 8 waves * 16x128 f16 (32 KB)

  const int qb = blockIdx.x, bh = blockIdx.y, tid = threadIdx.x;
  const int wave = tid >> 5, lane = tid & 31, grp = lane >> 4;
  const size_t base = (size_t)bh * N_ * D_ + (size_t)qb * 128 * D_;
  const size_t lb   = (size_t)bh * M_ * D_;

  if (wave == 0) {
    tdm_load_2d_f32(lds_off_bytes(sQf), Qg + base, D_, 128, D_, D_, N_);
    tdm_wait();
  }
  for (int i = tid; i < M_ * D_; i += 256) {
    sNC[i] = nc16[lb + i];
    sW[i]  = w16[lb + i];
  }
  __syncthreads();

  const int r0 = wave * 16;
  v8f cacc[8];
  for (int ct = 0; ct < 8; ++ct) {
    v8f a_ = v8f_zero();
#pragma unroll
    for (int kk = 0; kk < 2; ++kk) {
      v16h a = load_a16_f32(sQf, D_, r0, kk * 32);
      v16h b = load_b_col(sNC, D_, kk * 32, ct * 16);
      a_ = wmma_f16(a, b, a_);
    }
    cacc[ct] = a_;
  }
  float inv_l[8];
#pragma unroll
  for (int v = 0; v < 8; ++v) {
    float mx = -1e30f;
#pragma unroll
    for (int ct = 0; ct < 8; ++ct) mx = fmaxf(mx, cacc[ct][v]);
    mx = fmaxf(mx, __shfl_xor(mx, 1));
    mx = fmaxf(mx, __shfl_xor(mx, 2));
    mx = fmaxf(mx, __shfl_xor(mx, 4));
    mx = fmaxf(mx, __shfl_xor(mx, 8));
    float s = 0.0f;
#pragma unroll
    for (int ct = 0; ct < 8; ++ct) {
      const float e = __expf(cacc[ct][v] - mx);
      cacc[ct][v] = e; s += e;
    }
    s += __shfl_xor(s, 1); s += __shfl_xor(s, 2);
    s += __shfl_xor(s, 4); s += __shfl_xor(s, 8);
    inv_l[v] = 1.0f / s;
  }
  _Float16* sPw = sP + wave * 16 * 128;
  for (int ct = 0; ct < 8; ++ct)
#pragma unroll
    for (int v = 0; v < 8; ++v)
      sPw[(v + grp * 8) * 128 + ct * 16 + (lane & 15)] = (_Float16)cacc[ct][v];
  __syncthreads();

  for (int dt = 0; dt < 4; ++dt) {
    v8f acc = v8f_zero();
#pragma unroll
    for (int kk = 0; kk < 4; ++kk) {
      v16h a = load_a16(sPw, 128, 0, kk * 32);
      v16h b = load_b_row(sW, D_, kk * 32, dt * 16);
      acc = wmma_f16(a, b, acc);
    }
#pragma unroll
    for (int v = 0; v < 8; ++v) {
      const int lrow = v + grp * 8;
      Out[base + (r0 + lrow) * D_ + dt * 16 + (lane & 15)] = acc[v] * inv_l[lrow];
    }
  }
}

// ---------------------------------------------------------------------------
// Host launcher
// ---------------------------------------------------------------------------
extern "C" void kernel_launch(void* const* d_in, const int* in_sizes, int n_in,
                              void* d_out, int out_size, void* d_ws, size_t ws_size,
                              hipStream_t stream) {
  (void)in_sizes; (void)n_in; (void)out_size; (void)ws_size;
  const float* Q = (const float*)d_in[0];
  const float* K = (const float*)d_in[1];
  const float* V = (const float*)d_in[2];
  float* Out = (float*)d_out;

  char* p = (char*)d_ws;
  int* idxK = (int*)p;            p += (size_t)BHc * M_ * sizeof(int);
  int* idxQ = (int*)p;            p += (size_t)BHc * M_ * sizeof(int);
  float* a3v_part = (float*)p;    p += (size_t)BHc * S_ * M_ * D_ * sizeof(float);
  float* m_part = (float*)p;      p += (size_t)BHc * S_ * M_ * sizeof(float);
  float* l_part = (float*)p;      p += (size_t)BHc * S_ * M_ * sizeof(float);
  _Float16* nc16 = (_Float16*)p;  p += (size_t)BHc * M_ * D_ * sizeof(_Float16);
  _Float16* nr16 = (_Float16*)p;  p += (size_t)BHc * M_ * D_ * sizeof(_Float16);
  _Float16* k2i16 = (_Float16*)p; p += (size_t)BHc * M_ * M_ * sizeof(_Float16);
  _Float16* a3v16 = (_Float16*)p; p += (size_t)BHc * M_ * D_ * sizeof(_Float16);
  _Float16* w16 = (_Float16*)p;   p += (size_t)BHc * M_ * D_ * sizeof(_Float16);

  topk_kernel<<<dim3(BHc, 2), 256, 0, stream>>>(K, Q, idxK, idxQ);
  gather_kernel<<<BHc, 256, 0, stream>>>(K, Q, idxK, idxQ, nc16, nr16);

  inv_kernel<<<BHc, 256, (size_t)5 * 128 * 128 * sizeof(_Float16), stream>>>(
      nr16, nc16, k2i16);

  rkv_kernel<<<dim3(S_, BHc), 256, 65536, stream>>>(K, V, nr16, a3v_part,
                                                    m_part, l_part);
  merge_kernel<<<BHc, 256, 0, stream>>>(a3v_part, m_part, l_part, a3v16);

  wmat_kernel<<<BHc, 256, 49152, stream>>>(k2i16, a3v16, w16);

  out_kernel<<<dim3(N_ / 128, BHc), 256, 98304, stream>>>(Q, nc16, w16, Out);
}
